// Decoder_76974403879078
// MI455X (gfx1250) — compile-verified, matrix operand order checked by reading, not run on hardware
//
#include <hip/hip_runtime.h>
#include <stdint.h>

// ---------------------------------------------------------------------------
// MI455X / gfx1250 LSTM decoder.
//   - 16 workgroups x 256 threads (8 waves of 32); each WG owns 16 batch rows
//     and runs the entire 64-step scan locally (rows are independent).
//   - All matmuls (LSTM gates, cond, fc2, fc3) via v_wmma_f32_16x16x32_bf16;
//     weights pre-swizzled once per launch into WMMA B-fragment layout (bf16)
//     in the workspace (L2-resident, ~8.7 MB).
//   - Gate rows are interleaved (i,f,g,o per hidden unit) during the swizzle
//     so the LSTM pointwise update is wave-local (no big gate buffer).
//   - fc3 padded 33 -> 48 columns so it runs on the matrix pipe (waves 0-2).
// ---------------------------------------------------------------------------

typedef __attribute__((ext_vector_type(16))) __bf16   v16bf;
typedef __attribute__((ext_vector_type(8)))  float    v8f;
typedef __attribute__((ext_vector_type(4)))  unsigned u32x4;

union ABfrag { u32x4 q[2]; v16bf v; };

#define HID    512
#define NLAY   2
#define STEPS  64
#define OUTN   33
#define CONDK  128
#define ROWS   16     // batch rows per workgroup
#define NWG    16     // 256 / ROWS

// ---- dynamic LDS layout (bytes) -------------------------------------------
#define SMEM_HB    0            // bf16 h[2][16][512]           32768
#define SMEM_CB    32768        // f32  c[2][16][512]           65536
#define SMEM_XB    98304        // u32  x[16][256] bf16 pairs   16384
#define SMEM_COND  114688       // f32  cond[16][512]           32768
#define SMEM_YB    147456       // bf16 y[16][512]              16384
#define SMEM_BIAS  163840       // f32  bias[2][2048] permuted  16384
#define SMEM_SCR   180224       // f32  scratch[8][16][16]       8192
#define SMEM_PRED  188416       // f32  pred[16][33] (pad)       2176
#define SMEM_DURL  190592       // f32  durlogit[16][64]         4096
#define SMEM_RMAX  194688       // f32  [16]                       64
#define SMEM_RLSE  194752       // f32  [16]                       64
#define SMEM_ACT   194816       // int  [16]                       64
#define SMEM_TOTAL 194880

// ---- helpers --------------------------------------------------------------
__device__ __forceinline__ unsigned short f32_bf16(float f) {
  unsigned u = __float_as_uint(f);
  u += 0x7FFFu + ((u >> 16) & 1u);          // round-to-nearest-even
  return (unsigned short)(u >> 16);
}
__device__ __forceinline__ unsigned pack2_bf16(float a, float b) {
  return (unsigned)f32_bf16(a) | ((unsigned)f32_bf16(b) << 16);
}
__device__ __forceinline__ float bf16_f32(unsigned short h) {
  return __uint_as_float(((unsigned)h) << 16);
}
__device__ __forceinline__ float sigm(float x) {
  return 1.0f / (1.0f + __expf(-x));
}
__device__ __forceinline__ float lrelu(float x) {
  return x >= 0.0f ? x : 0.01f * x;
}

// A fragment (16x32 bf16, M rows in LDS, row stride rs u32): lane L holds row
// m=L&15; K pairs at k = kt*32 + (j>>2)*16 + (j&3)*2 + (L>>4)*8  (ISA 7.12.2).
// Pairs are contiguous u32s -> two ds_load_b128 per fragment.
__device__ __forceinline__ v16bf load_afrag(const unsigned* base, int rs,
                                            int kt, int lane) {
  const int m = lane & 15, half = lane >> 4;
  const int b0 = m * rs + kt * 16 + half * 4;
  ABfrag r;
  r.q[0] = *(const u32x4*)(base + b0);
  r.q[1] = *(const u32x4*)(base + b0 + 8);
  return r.v;
}
// B fragment pre-swizzled in global memory: 32 lanes x 8 u32 per fragment.
__device__ __forceinline__ v16bf load_bfrag(const unsigned* __restrict__ w,
                                            int frag, int lane) {
  const unsigned* p = w + ((size_t)frag * 32 + lane) * 8;
  ABfrag r;
  r.q[0] = *(const u32x4*)p;
  r.q[1] = *(const u32x4*)(p + 4);
  return r.v;
}
__device__ __forceinline__ v8f wmma_bf16(v16bf a, v16bf b, v8f c) {
  return __builtin_amdgcn_wmma_f32_16x16x32_bf16(false, a, false, b,
                                                 (short)0, c, false, false);
}

// ---------------------------------------------------------------------------
// Weight pre-swizzle kernels: f32 row-major -> bf16 WMMA B-fragments.
// Fragment (n_tile, kt): out[((n_tile*KT+kt)*32 + lane)*8 + j] packs W[n][k],
// W[n][k+1] with n = n_tile*16 + (lane&15),
// k = kt*32 + (j>>2)*16 + (j&3)*2 + (lane>>4)*8.
// ---------------------------------------------------------------------------

// LSTM layer: Kc = 1024 (Wih | Whh concatenated), KT=32, N=2048, rows
// permuted so new row n maps to torch row (n&3)*512 + (n>>2)  (gate interleave).
__global__ void k_swizzle_lstm(const float* __restrict__ Wih,
                               const float* __restrict__ Whh,
                               unsigned* __restrict__ out) {
  int tid  = blockIdx.x * blockDim.x + threadIdx.x;  // one u32 pair each
  int j    = tid & 7;
  int lane = (tid >> 3) & 31;
  int fr   = tid >> 8;
  int kt   = fr & 31, nt = fr >> 5;
  int n    = nt * 16 + (lane & 15);
  int src  = (n & 3) * 512 + (n >> 2);               // gate-permuted row
  int k    = kt * 32 + (j >> 2) * 16 + (j & 3) * 2 + (lane >> 4) * 8;
  float a, b;
  if (k < HID) { a = Wih[(size_t)src * HID + k];
                 b = Wih[(size_t)src * HID + k + 1]; }
  else         { a = Whh[(size_t)src * HID + (k - HID)];
                 b = Whh[(size_t)src * HID + (k - HID) + 1]; }
  out[tid] = pack2_bf16(a, b);
}

// Plain matrix [N][K] row-major, no permutation.
__global__ void k_swizzle_plain(const float* __restrict__ W,
                                unsigned* __restrict__ out, int K, int KT) {
  int tid  = blockIdx.x * blockDim.x + threadIdx.x;
  int j    = tid & 7;
  int lane = (tid >> 3) & 31;
  int fr   = tid >> 8;
  int kt   = fr % KT, nt = fr / KT;
  int n    = nt * 16 + (lane & 15);
  int k    = kt * 32 + (j >> 2) * 16 + (j & 3) * 2 + (lane >> 4) * 8;
  out[tid] = pack2_bf16(W[(size_t)n * K + k], W[(size_t)n * K + k + 1]);
}

// fc3 [33][512] padded to 48 rows (zeros) so pred runs on the matrix pipe.
__global__ void k_swizzle_fc3(const float* __restrict__ W,
                              unsigned* __restrict__ out) {
  int tid  = blockIdx.x * blockDim.x + threadIdx.x;  // 48 frags * 256
  int j    = tid & 7;
  int lane = (tid >> 3) & 31;
  int fr   = tid >> 8;
  int kt   = fr & 15, nt = fr >> 4;                  // KT = 16
  int n    = nt * 16 + (lane & 15);
  int k    = kt * 32 + (j >> 2) * 16 + (j & 3) * 2 + (lane >> 4) * 8;
  float a = 0.0f, b = 0.0f;
  if (n < OUTN) { a = W[(size_t)n * HID + k]; b = W[(size_t)n * HID + k + 1]; }
  out[tid] = pack2_bf16(a, b);
}

// ---------------------------------------------------------------------------
// Main persistent-per-workgroup decoder kernel.
// ---------------------------------------------------------------------------
__global__ __launch_bounds__(256)
void k_decoder(const float* __restrict__ h0,  const float* __restrict__ c0,
               const float* __restrict__ cnd, const float* __restrict__ emb,
               const float* __restrict__ bih, const float* __restrict__ bhh,
               const float* __restrict__ fcb, const float* __restrict__ fc2b,
               const float* __restrict__ fc3b,
               const unsigned* __restrict__ w0, const unsigned* __restrict__ w1,
               const unsigned* __restrict__ wfc2,
               const unsigned* __restrict__ wfcc,
               const unsigned* __restrict__ wfc3,
               float* __restrict__ out) {
  extern __shared__ char smem[];
  unsigned short* hb  = (unsigned short*)(smem + SMEM_HB);   // bf16 h state
  unsigned*       hb32= (unsigned*)      (smem + SMEM_HB);   // u32 view
  float*          cb  = (float*)(smem + SMEM_CB);
  unsigned*       xb  = (unsigned*)(smem + SMEM_XB);
  float*          cdb = (float*)(smem + SMEM_COND);
  unsigned short* yb16= (unsigned short*)(smem + SMEM_YB);
  unsigned*       yb32= (unsigned*)      (smem + SMEM_YB);
  float*          bsb = (float*)(smem + SMEM_BIAS);
  float*          scr = (float*)(smem + SMEM_SCR);
  float*          prd = (float*)(smem + SMEM_PRED);
  float*          dur = (float*)(smem + SMEM_DURL);
  float*          rmx = (float*)(smem + SMEM_RMAX);
  float*          rls = (float*)(smem + SMEM_RLSE);
  int*            act = (int*)(smem + SMEM_ACT);

  const int t = threadIdx.x, wave = t >> 5, lane = t & 31;
  const int wg = blockIdx.x;
  const int col = lane & 15, half = lane >> 4;

  // ---- init: h/c state, permuted bias sums, SOS token -----------------
  for (int q = 0; q < 64; ++q) {             // 2*16*512 elems / 256 threads
    int idx = t + 256 * q;
    int l = idx >> 13, m = (idx >> 9) & 15, u = idx & 511;
    size_t g = ((size_t)l * 256 + wg * ROWS + m) * HID + u;
    hb[idx] = f32_bf16(h0[g]);
    cb[idx] = c0[g];
  }
  for (int q = 0; q < 16; ++q) {             // 2*2048 / 256
    int idx = t + 256 * q;
    int l = idx >> 11, nn = idx & 2047;
    int src = (nn & 3) * 512 + (nn >> 2);    // same gate permutation
    bsb[idx] = bih[l * 2048 + src] + bhh[l * 2048 + src];
  }
  if (t < ROWS) act[t] = 0;                  // SOS
  // stage conditionals as bf16 pairs into xb ([16][64] u32)
  for (int q = 0; q < 4; ++q) {
    int p = t + 256 * q;
    int m = p >> 6, kp = p & 63;
    const float* s = cnd + ((size_t)(wg * ROWS + m)) * CONDK + kp * 2;
    xb[p] = pack2_bf16(s[0], s[1]);
  }
  __syncthreads();

  // ---- cond = leaky(conditionals @ fcW^T + fcb), once -----------------
  {
    v8f acc[4];
#pragma unroll
    for (int i = 0; i < 4; ++i) acc[i] = (v8f){0,0,0,0,0,0,0,0};
    for (int kt = 0; kt < 4; ++kt) {
      v16bf a = load_afrag(xb, 64, kt, lane);
#pragma unroll
      for (int i = 0; i < 4; ++i) {
        v16bf b = load_bfrag(wfcc, (wave * 4 + i) * 4 + kt, lane);
        acc[i] = wmma_bf16(a, b, acc[i]);
      }
    }
#pragma unroll
    for (int i = 0; i < 4; ++i) {
      int n = (wave * 4 + i) * 16 + col;
      float bv = fcb[n];
#pragma unroll
      for (int r = 0; r < 8; ++r)
        cdb[(half * 8 + r) * HID + n] = lrelu(acc[i][r] + bv);
    }
  }
  __syncthreads();

  // =====================  64-step sequential scan  =====================
  for (int step = 0; step < STEPS; ++step) {
    // ---- x0 = [emb[act], dur] as bf16 pairs in xb ----
    float dv = (step == 0) ? 0.0f : 1.0f;
    for (int q = 0; q < 16; ++q) {
      int p = t + 256 * q;
      int m = p >> 8, kp = p & 255, k0 = kp * 2;
      int a = act[m];
      float f0 = (k0 < HID - 1) ? emb[(size_t)a * (HID - 1) + k0] : dv;
      float f1 = (k0 + 1 < HID - 1) ? emb[(size_t)a * (HID - 1) + k0 + 1] : dv;
      xb[p] = pack2_bf16(f0, f1);
    }
    __syncthreads();

    // ---- two LSTM layers ----
    for (int l = 0; l < NLAY; ++l) {
      const unsigned* xsrc = (l == 0) ? xb : hb32;      // layer1 x = new h0
      const unsigned* hsrc = hb32 + l * 4096;           // 16*256 u32 per layer
      const unsigned* wsw  = (l == 0) ? w0 : w1;
      v8f acc[16];
#pragma unroll
      for (int i = 0; i < 16; ++i) acc[i] = (v8f){0,0,0,0,0,0,0,0};
      for (int kt = 0; kt < 32; ++kt) {                 // Kc = 1024
        v16bf a = (kt < 16) ? load_afrag(xsrc, 256, kt, lane)
                            : load_afrag(hsrc, 256, kt - 16, lane);
#pragma unroll
        for (int i = 0; i < 16; ++i) {
          v16bf b = load_bfrag(wsw, (wave * 16 + i) * 32 + kt, lane);
          acc[i] = wmma_bf16(a, b, acc[i]);
        }
      }
      __syncthreads();   // every wave done reading h[l] / x fragments

      // pointwise LSTM update: tile cols = 4 units x 4 interleaved gates
      float* ms = scr + wave * 256;
#pragma unroll
      for (int i = 0; i < 16; ++i) {
        int nt = wave * 16 + i;
        float bv = bsb[l * 2048 + nt * 16 + col];
#pragma unroll
        for (int r = 0; r < 8; ++r)
          ms[(half * 8 + r) * 16 + col] = acc[i][r] + bv;
        __builtin_amdgcn_wave_barrier();                // wave-local exchange
#pragma unroll
        for (int e = 0; e < 2; ++e) {
          int idx = lane * 2 + e;
          int m = idx >> 2, uu = idx & 3;
          float ig = ms[m * 16 + uu * 4 + 0];
          float fg = ms[m * 16 + uu * 4 + 1];
          float gg = ms[m * 16 + uu * 4 + 2];
          float og = ms[m * 16 + uu * 4 + 3];
          int u = nt * 4 + uu;
          int ci = (l * ROWS + m) * HID + u;
          float cn = sigm(fg) * cb[ci] + sigm(ig) * tanhf(gg);
          cb[ci] = cn;
          hb[ci] = f32_bf16(sigm(og) * tanhf(cn));
        }
        __builtin_amdgcn_wave_barrier();
      }
      __syncthreads();   // h[l] fully updated before next consumer
    }

    // ---- z = h1 + cond (bf16 pairs into xb) ----
    for (int q = 0; q < 16; ++q) {
      int p = t + 256 * q;
      int m = p >> 8, k0 = (p & 255) * 2;
      float z0 = bf16_f32(hb[(ROWS + m) * HID + k0]) + cdb[m * HID + k0];
      float z1 = bf16_f32(hb[(ROWS + m) * HID + k0 + 1]) + cdb[m * HID + k0 + 1];
      xb[p] = pack2_bf16(z0, z1);
    }
    __syncthreads();

    // ---- y = leaky(z @ fc2W^T + fc2b), stored as bf16 for fc3 ----
    {
      v8f acc[4];
#pragma unroll
      for (int i = 0; i < 4; ++i) acc[i] = (v8f){0,0,0,0,0,0,0,0};
      for (int kt = 0; kt < 16; ++kt) {
        v16bf a = load_afrag(xb, 256, kt, lane);
#pragma unroll
        for (int i = 0; i < 4; ++i) {
          v16bf b = load_bfrag(wfc2, (wave * 4 + i) * 16 + kt, lane);
          acc[i] = wmma_bf16(a, b, acc[i]);
        }
      }
#pragma unroll
      for (int i = 0; i < 4; ++i) {
        int n = (wave * 4 + i) * 16 + col;
        float bv = fc2b[n];
#pragma unroll
        for (int r = 0; r < 8; ++r)
          yb16[(half * 8 + r) * HID + n] = f32_bf16(lrelu(acc[i][r] + bv));
      }
    }
    __syncthreads();

    // ---- pred = y @ fc3W^T + fc3b on the matrix pipe (waves 0-2) ----
    if (wave < 3) {                          // wave-uniform: EXEC all-ones
      v8f acc = (v8f){0,0,0,0,0,0,0,0};
      for (int kt = 0; kt < 16; ++kt) {
        v16bf a = load_afrag(yb32, 256, kt, lane);
        v16bf b = load_bfrag(wfc3, wave * 16 + kt, lane);
        acc = wmma_bf16(a, b, acc);
      }
      int n = wave * 16 + col;
      if (n < OUTN) {
        float bv = fc3b[n];
#pragma unroll
        for (int r = 0; r < 8; ++r)
          prd[(half * 8 + r) * OUTN + n] = acc[r] + bv;
      }
    }
    __syncthreads();

    // ---- argmax + log_softmax prep (per row) ----
    if (t < ROWS) {
      float mx = -1e30f; int am = 0;
      for (int o = 0; o < OUTN - 1; ++o) {
        float v = prd[t * OUTN + o];
        if (v > mx) { mx = v; am = o; }
      }
      float s = 0.0f;
      for (int o = 0; o < OUTN - 1; ++o) s += __expf(prd[t * OUTN + o] - mx);
      rmx[t] = mx; rls[t] = __logf(s); act[t] = am;
      dur[t * STEPS + step] = prd[t * OUTN + (OUTN - 1)];
    }
    __syncthreads();

    // ---- write activity log-probs for this step ----
    {
      int m = t & 15, o = t >> 4;   // o in 0..15, handles o and o+16
      size_t base = ((size_t)(wg * ROWS + m) * STEPS + step) * OUTN;
      float sub = rmx[m] + rls[m];
      out[base + o]      = prd[m * OUTN + o]      - sub;
      out[base + o + 16] = prd[m * OUTN + o + 16] - sub;
    }
    __syncthreads();
  }

  // ---- durations: softmax over time per row ----
  if (t < ROWS) {
    float mx = -1e30f;
    for (int s2 = 0; s2 < STEPS; ++s2) mx = fmaxf(mx, dur[t * STEPS + s2]);
    float s = 0.0f;
    for (int s2 = 0; s2 < STEPS; ++s2) s += __expf(dur[t * STEPS + s2] - mx);
    rmx[t] = mx; rls[t] = 1.0f / s;
  }
  __syncthreads();
  {
    int m = t & 15;
    for (int q = 0; q < 4; ++q) {
      int tt = (t >> 4) + q * 16;
      out[((size_t)(wg * ROWS + m) * STEPS + tt) * OUTN + (OUTN - 1)] =
          __expf(dur[m * STEPS + tt] - rmx[m]) * rls[m];
    }
  }
}

// ---------------------------------------------------------------------------
extern "C" void kernel_launch(void* const* d_in, const int* in_sizes, int n_in,
                              void* d_out, int out_size, void* d_ws,
                              size_t ws_size, hipStream_t stream) {
  const float* h0   = (const float*)d_in[1];
  const float* c0   = (const float*)d_in[2];
  const float* cnd  = (const float*)d_in[3];
  const float* emb  = (const float*)d_in[4];
  const float* Wih  = (const float*)d_in[5];
  const float* Whh  = (const float*)d_in[6];
  const float* bih  = (const float*)d_in[7];
  const float* bhh  = (const float*)d_in[8];
  const float* fcW  = (const float*)d_in[9];
  const float* fcb  = (const float*)d_in[10];
  const float* fc2W = (const float*)d_in[11];
  const float* fc2b = (const float*)d_in[12];
  const float* fc3W = (const float*)d_in[13];
  const float* fc3b = (const float*)d_in[14];
  float* out = (float*)d_out;

  // workspace: bf16 weights in WMMA fragment layout (~8.7 MB, L2-resident)
  unsigned* w0   = (unsigned*)d_ws;          // 2048*1024/2 u32
  unsigned* w1   = w0 + 1048576;
  unsigned* wfc2 = w1 + 1048576;             // 512*512/2
  unsigned* wfcc = wfc2 + 131072;            // 512*128/2
  unsigned* wfc3 = wfcc + 32768;             // 48*512/2 (padded)
  (void)in_sizes; (void)n_in; (void)out_size; (void)ws_size;

  k_swizzle_lstm<<<4096, 256, 0, stream>>>(Wih,              Whh,              w0);
  k_swizzle_lstm<<<4096, 256, 0, stream>>>(Wih + 2048 * 512, Whh + 2048 * 512, w1);
  k_swizzle_plain<<<512, 256, 0, stream>>>(fc2W, wfc2, 512, 16);
  k_swizzle_plain<<<128, 256, 0, stream>>>(fcW,  wfcc, 128, 4);
  k_swizzle_fc3 <<<48,  256, 0, stream>>>(fc3W, wfc3);

  hipFuncSetAttribute((const void*)k_decoder,
                      hipFuncAttributeMaxDynamicSharedMemorySize, SMEM_TOTAL);
  k_decoder<<<NWG, 256, SMEM_TOTAL, stream>>>(
      h0, c0, cnd, emb, bih, bhh, fcb, fc2b, fc3b,
      w0, w1, wfc2, wfcc, wfc3, out);
}
